// GLOM_encoder_17085379903797
// MI455X (gfx1250) — matrix operand order repaired and checked by reading, not compiled, not run on hardware
//
#include <hip/hip_runtime.h>
#include <hip/hip_bf16.h>

// ---------------------------------------------------------------------------
// GLOM encoder for MI455X (gfx1250, wave32, WMMA).
// Dominant cost: td_w (12800x12900) and bu_w (12800x12800) GEMMs, batched to
// M=64 rows (4 levels x B=16). Weights streamed fp32 from HBM once per step
// (roofline floor ~3.95 GB over 3 steps => ~170us at 23.3 TB/s), converted to
// bf16 in registers, consumed by v_wmma_f32_16x16x32_bf16.
// K dimension split 8 ways per strip (6400 waves) so in-flight load bytes
// cover the HBM bandwidth-delay product; partials reduced deterministically.
// readfirstlane makes the per-wave K-range provably uniform so the hot loop
// stays a scalar-bounded, branch-free, EXEC=all-ones loop.
// ---------------------------------------------------------------------------

typedef __attribute__((ext_vector_type(16))) __bf16 bf16x16;
typedef __attribute__((ext_vector_type(8)))  __bf16 bf16x8;
typedef __attribute__((ext_vector_type(8)))  float  f32x8;
typedef __attribute__((ext_vector_type(4)))  float  f32x4;

namespace {
constexpr int BB  = 16;     // batch
constexpr int P   = 100;    // positions
constexpr int E   = 5;      // levels
constexpr int D   = 128;    // dim
constexpr int CG  = 10;     // P / G
constexpr int HID = 32;
constexpr int PD   = P * D;        // 12800
constexpr int KTD  = PD + P;       // 12900 (top-down input width)
constexpr int KTDP = 12928;        // padded to multiple of 32
constexpr int MROWS = 64;          // 4 levels x 16 batch rows
constexpr int BPED  = BB * P * E * D;
constexpr int NSPLIT = 8;          // K-split factor for the big GEMMs
}

// ---------------- CNN (3x3 patches) + FC, broadcast to all levels ----------
__global__ void k_init(const float* __restrict__ x, const float* __restrict__ cw,
                       const float* __restrict__ cb, const float* __restrict__ fw,
                       const float* __restrict__ fb, float* __restrict__ cols)
{
    int bp = blockIdx.x;            // b*P + p
    int b = bp / P, p = bp % P;
    int py = p / 10, px = p % 10;
    __shared__ float h[HID];
    int t = threadIdx.x;
    if (t < HID) {
        float s = cb[t];
        #pragma unroll
        for (int i = 0; i < 3; ++i) {
            int r = py * 3 + i - 1;
            #pragma unroll
            for (int j = 0; j < 3; ++j) {
                int c = px * 3 + j - 1;
                float v = (r >= 0 && r < 28 && c >= 0 && c < 28)
                        ? x[(b * 28 + r) * 28 + c] : 0.0f;
                s += v * cw[t * 9 + i * 3 + j];
            }
        }
        h[t] = fmaxf(s, 0.0f);
    }
    __syncthreads();
    float e = fb[t];
    #pragma unroll
    for (int o = 0; o < HID; ++o) e += h[o] * fw[t * HID + o];
    #pragma unroll
    for (int lvl = 0; lvl < E; ++lvl)
        cols[((size_t)bp * E + lvl) * D + t] = e;
}

// ---------------- grouped conv producing q,k,v for all levels --------------
__global__ void k_gconv(const float* __restrict__ cols,
                        const float* __restrict__ qw, const float* __restrict__ qb,
                        const float* __restrict__ kw, const float* __restrict__ kb,
                        const float* __restrict__ vw, const float* __restrict__ vb,
                        float* __restrict__ q, float* __restrict__ k,
                        float* __restrict__ v)
{
    int idx = blockIdx.x * blockDim.x + threadIdx.x;
    if (idx >= E * BB * P * D) return;
    int d  = idx % D;
    int pp = (idx / D) % P;
    int b  = (idx / (D * P)) % BB;
    int l  =  idx / (D * P * BB);
    int g = pp / CG, o = pp % CG;
    float sq = 0.f, sk = 0.f, sv = 0.f;
    #pragma unroll
    for (int c = 0; c < CG; ++c) {
        float z = cols[(((size_t)(b * P + g * CG + c) * E) + l) * D + d];
        int wi = (g * CG + o) * CG + c;
        sq += qw[wi] * z;
        sk += kw[wi] * z;
        sv += vw[wi] * z;
    }
    size_t out = ((size_t)(l * BB + b) * P + pp) * D + d;
    q[out] = sq + qb[pp];
    k[out] = sk + kb[pp];
    v[out] = sv + vb[pp];
}

// ---------------- S[l,b,n,m] = sum_p q[p,n]*k[p,m] -------------------------
__global__ void k_scores(const float* __restrict__ q, const float* __restrict__ k,
                         float* __restrict__ S)
{
    int lb = blockIdx.x;
    const float* qp = q + (size_t)lb * P * D;
    const float* kp = k + (size_t)lb * P * D;
    float* Sp = S + (size_t)lb * D * D;
    for (int pair = threadIdx.x; pair < D * D; pair += blockDim.x) {
        int n = pair / D, m = pair % D;
        float s = 0.f;
        for (int p = 0; p < P; ++p)
            s += qp[p * D + n] * kp[p * D + m];
        Sp[pair] = s;
    }
}

// ---------------- row softmax over last dim (128) --------------------------
__global__ void k_softmax(float* __restrict__ S)
{
    __shared__ float red[D];
    float* row = S + (size_t)blockIdx.x * D;
    int t = threadIdx.x;
    float val = row[t];
    red[t] = val;
    __syncthreads();
    for (int off = 64; off > 0; off >>= 1) {
        if (t < off) red[t] = fmaxf(red[t], red[t + off]);
        __syncthreads();
    }
    float mx = red[0];
    __syncthreads();
    float ex = __expf(val - mx);
    red[t] = ex;
    __syncthreads();
    for (int off = 64; off > 0; off >>= 1) {
        if (t < off) red[t] += red[t + off];
        __syncthreads();
    }
    row[t] = ex / red[0];
}

// ---------------- ta = gamma * (v @ A^T) + t1 ------------------------------
__global__ void k_attnout(const float* __restrict__ v, const float* __restrict__ S,
                          const float* __restrict__ cols,
                          const float* __restrict__ gamma,
                          float* __restrict__ ta)
{
    int blk = blockIdx.x;           // (l*BB+b)*P + p
    int p  = blk % P;
    int lb = blk / P;
    int b = lb % BB, l = lb / BB;
    __shared__ float vs[D];
    int n = threadIdx.x;
    vs[n] = v[((size_t)lb * P + p) * D + n];
    __syncthreads();
    const float* Srow = S + ((size_t)lb * D + n) * D;
    float acc = 0.f;
    #pragma unroll 4
    for (int m = 0; m < D; ++m) acc += vs[m] * Srow[m];
    float t1 = cols[((size_t)(b * P + p) * E + l) * D + n];
    ta[((size_t)lb * P + p) * D + n] = gamma[0] * acc + t1;
}

// ---------------- pack activations to bf16 GEMM inputs ---------------------
__global__ void k_pack_td(const float* __restrict__ cols, __bf16* __restrict__ Xtd)
{
    int idx = blockIdx.x * blockDim.x + threadIdx.x;
    if (idx >= MROWS * KTDP) return;
    int r = idx / KTDP, i = idx % KTDP;
    int lvl = r >> 4, b = r & 15;   // source level lvl+1 (target level lvl)
    float val;
    if (i < PD) {
        int p = i / D, d = i % D;
        val = cols[((size_t)(b * P + p) * E + (lvl + 1)) * D + d];
    } else if (i < KTD) {
        val = (float)(i - PD);      // pos vector
    } else {
        val = 0.0f;                 // zero pad to K multiple of 32
    }
    Xtd[idx] = (__bf16)val;
}

__global__ void k_pack_bu(const float* __restrict__ cols, __bf16* __restrict__ Xbu)
{
    int idx = blockIdx.x * blockDim.x + threadIdx.x;
    if (idx >= MROWS * PD) return;
    int r = idx / PD, i = idx % PD;
    int lvl = r >> 4, b = r & 15;   // source level lvl (target level lvl+1)
    int p = i / D, d = i % D;
    Xbu[idx] = (__bf16)cols[((size_t)(b * P + p) * E + lvl) * D + d];
}

// ---------------- Ypart(s, 64 x Nout) = Xbf * W^T (K-chunk s) --------------
static __device__ __forceinline__ f32x8 wmma_bf16(bf16x16 a, bf16x16 b, f32x8 c)
{
    return __builtin_amdgcn_wmma_f32_16x16x32_bf16(false, a, false, b,
                                                   (short)0, c, false, false);
}

static __device__ __forceinline__ bf16x16 load_afrag(const __bf16* Xr, int ka0, int ka1)
{
    bf16x8 lo = *(const bf16x8*)(Xr + ka0);
    bf16x8 hi = *(const bf16x8*)(Xr + ka1);
    return __builtin_shufflevector(lo, hi,
        0,1,2,3,4,5,6,7,8,9,10,11,12,13,14,15);
}

__global__ void k_gemm64(const __bf16* __restrict__ Xbf, const float* __restrict__ W,
                         float* __restrict__ Ypart,
                         int Kpad, int Kw, int Nout)
{
    // readfirstlane: wave index is identical across the wave; forcing it into
    // an SGPR makes strip/split/kstart/kend provably uniform, so the K loop
    // compiles to a scalar-bounded loop (no EXEC masking around WMMA).
    int wid   = __builtin_amdgcn_readfirstlane(
                    (int)((blockIdx.x * blockDim.x + threadIdx.x) >> 5));
    int lane  = threadIdx.x & 31;
    int strip = wid >> 3;            // wid / NSPLIT
    int split = wid & (NSPLIT - 1);
    int nbase = strip * 16;
    if (nbase >= Nout) return;       // wave-uniform: EXEC all-ones at WMMA
    int half = lane >> 4;            // 0: lanes 0-15, 1: lanes 16-31
    int l15  = lane & 15;

    const float*  Wr = W + (size_t)(nbase + l15) * Kw;   // one W row per lane
    const __bf16* X0 = Xbf + (size_t)(0 * 16 + l15) * Kpad;
    const __bf16* X1 = Xbf + (size_t)(1 * 16 + l15) * Kpad;
    const __bf16* X2 = Xbf + (size_t)(2 * 16 + l15) * Kpad;
    const __bf16* X3 = Xbf + (size_t)(3 * 16 + l15) * Kpad;

    f32x8 acc0 = {}, acc1 = {}, acc2 = {}, acc3 = {};

    // This wave's K range (SGPR arithmetic after readfirstlane).
    int Kchunk = (Kpad / NSPLIT + 31) & ~31;
    int kstart = split * Kchunk;
    int kend   = min(kstart + Kchunk, Kpad);
    int Kfast  = Kw & ~31;                 // largest multiple of 32 inside W
    int fastEnd = min(kend, Kfast);

    #pragma unroll 2
    for (int kk = kstart; kk < fastEnd; kk += 32) {
        // B fragment: lane (half,l15) holds B[k, n=l15] = W[nbase+l15, k],
        // k in [kk + 16*half, +16). The two lane-halves together consume a
        // contiguous 128B line of the W row each iteration.
        int kb = kk + half * 16;
        __builtin_prefetch(Wr + kb + 256, 0, 1);         // global_prefetch_b8
        const f32x4* w4 = (const f32x4*)(Wr + kb);       // 16B aligned
        f32x4 w0 = w4[0], w1 = w4[1], w2 = w4[2], w3 = w4[3];
        bf16x16 bfrag;
        #pragma unroll
        for (int t = 0; t < 4; ++t) {
            bfrag[t]      = (__bf16)w0[t];
            bfrag[4 + t]  = (__bf16)w1[t];
            bfrag[8 + t]  = (__bf16)w2[t];
            bfrag[12 + t] = (__bf16)w3[t];
        }
        // A fragments (16-bit A 16x32 layout): lane row m=l15,
        // K = {kk+16*half .. +15} (contiguous, matching B)
        int ka0 = kk + half * 16;
        int ka1 = ka0 + 8;
        bf16x16 a0 = load_afrag(X0, ka0, ka1);
        bf16x16 a1 = load_afrag(X1, ka0, ka1);
        bf16x16 a2 = load_afrag(X2, ka0, ka1);
        bf16x16 a3 = load_afrag(X3, ka0, ka1);

        acc0 = wmma_bf16(a0, bfrag, acc0);
        acc1 = wmma_bf16(a1, bfrag, acc1);
        acc2 = wmma_bf16(a2, bfrag, acc2);
        acc3 = wmma_bf16(a3, bfrag, acc3);
    }

    // Ragged tail group (td only: exactly the split containing Kfast=12896
    // runs one guarded group up to Kpad=12928). Scalar branch, uniform.
    if (fastEnd < kend) {
        int kk = fastEnd;
        int kb = kk + half * 16;
        bf16x16 bfrag;
        #pragma unroll
        for (int t = 0; t < 16; ++t) {
            int c = kb + t;
            bfrag[t] = (__bf16)((c < Kw) ? Wr[c] : 0.0f);
        }
        int ka0 = kk + half * 16;
        int ka1 = ka0 + 8;                // A is zero-padded to Kpad: safe
        bf16x16 a0 = load_afrag(X0, ka0, ka1);
        bf16x16 a1 = load_afrag(X1, ka0, ka1);
        bf16x16 a2 = load_afrag(X2, ka0, ka1);
        bf16x16 a3 = load_afrag(X3, ka0, ka1);
        acc0 = wmma_bf16(a0, bfrag, acc0);
        acc1 = wmma_bf16(a1, bfrag, acc1);
        acc2 = wmma_bf16(a2, bfrag, acc2);
        acc3 = wmma_bf16(a3, bfrag, acc3);
    }

    // C/D layout: VGPR r -> M = r + 8*half, N = l15. Partial (no bias).
    float* Yp = Ypart + (size_t)split * MROWS * Nout;
    int col = nbase + l15;
    #pragma unroll
    for (int r = 0; r < 8; ++r) {
        int ml = r + half * 8;
        Yp[(size_t)(0 * 16 + ml) * Nout + col] = acc0[r];
        Yp[(size_t)(1 * 16 + ml) * Nout + col] = acc1[r];
        Yp[(size_t)(2 * 16 + ml) * Nout + col] = acc2[r];
        Yp[(size_t)(3 * 16 + ml) * Nout + col] = acc3[r];
    }
}

// ---------------- Y = sum_s Ypart[s] + bias (fixed order: deterministic) ---
__global__ void k_reduce(const float* __restrict__ parts,
                         const float* __restrict__ bias,
                         float* __restrict__ Y, int Nout)
{
    int idx = blockIdx.x * blockDim.x + threadIdx.x;
    if (idx >= MROWS * Nout) return;
    int col = idx % Nout;
    float s = bias[col];
    #pragma unroll
    for (int sp = 0; sp < NSPLIT; ++sp)
        s += parts[(size_t)sp * MROWS * Nout + idx];
    Y[idx] = s;
}

// ---------------- combine: new = t1 + .33*td + .33*bu + .33*ta -------------
__global__ void k_combine(const float* __restrict__ colsCur, const float* __restrict__ ta,
                          const float* __restrict__ Ytd, const float* __restrict__ Ybu,
                          float* __restrict__ colsNext)
{
    int idx = blockIdx.x * blockDim.x + threadIdx.x;
    if (idx >= BPED) return;
    int d = idx % D;
    int l = (idx / D) % E;
    int p = (idx / (D * E)) % P;
    int b =  idx / (D * E * P);
    float acc = colsCur[idx]
              + 0.33f * ta[((size_t)(l * BB + b) * P + p) * D + d];
    if (l < E - 1) acc += 0.33f * Ytd[(size_t)(l * 16 + b) * PD + p * D + d];
    if (l > 0)     acc += 0.33f * Ybu[(size_t)((l - 1) * 16 + b) * PD + p * D + d];
    colsNext[idx] = acc;
}

// ---------------- out[b,d] = mean_p cols[b,p,E-1,d] ------------------------
__global__ void k_final(const float* __restrict__ cols, float* __restrict__ out)
{
    int idx = blockIdx.x * blockDim.x + threadIdx.x;
    if (idx >= BB * D) return;
    int b = idx / D, d = idx % D;
    float s = 0.f;
    for (int p = 0; p < P; ++p)
        s += cols[((size_t)(b * P + p) * E + (E - 1)) * D + d];
    out[idx] = s * (1.0f / (float)P);
}

// ---------------------------------------------------------------------------
extern "C" void kernel_launch(void* const* d_in, const int* in_sizes, int n_in,
                              void* d_out, int out_size, void* d_ws, size_t ws_size,
                              hipStream_t stream)
{
    (void)in_sizes; (void)n_in; (void)out_size; (void)ws_size;
    const float* x     = (const float*)d_in[0];
    const float* cw    = (const float*)d_in[1];
    const float* cb    = (const float*)d_in[2];
    const float* fw    = (const float*)d_in[3];
    const float* fb    = (const float*)d_in[4];
    const float* tdw   = (const float*)d_in[5];
    const float* tdb   = (const float*)d_in[6];
    const float* buw   = (const float*)d_in[7];
    const float* bub   = (const float*)d_in[8];
    const float* qw    = (const float*)d_in[9];
    const float* qb    = (const float*)d_in[10];
    const float* kw    = (const float*)d_in[11];
    const float* kb    = (const float*)d_in[12];
    const float* vw    = (const float*)d_in[13];
    const float* vb    = (const float*)d_in[14];
    const float* gamma = (const float*)d_in[15];

    char* ws = (char*)d_ws;
    size_t off = 0;
    auto alloc = [&](size_t bytes) -> void* {
        off = (off + 255) & ~(size_t)255;
        void* p = ws + off;
        off += bytes;
        return p;
    };

    float*  cols0 = (float*)alloc((size_t)BPED * 4);
    float*  cols1 = (float*)alloc((size_t)BPED * 4);
    float*  qbuf  = (float*)alloc((size_t)BPED * 4);
    float*  kbuf  = (float*)alloc((size_t)BPED * 4);
    float*  vbuf  = (float*)alloc((size_t)BPED * 4);
    float*  Sbuf  = (float*)alloc((size_t)E * BB * D * D * 4);
    float*  tabuf = (float*)alloc((size_t)BPED * 4);
    __bf16* Xtd   = (__bf16*)alloc((size_t)MROWS * KTDP * 2);
    __bf16* Xbu   = (__bf16*)alloc((size_t)MROWS * PD * 2);
    float*  Ytd   = (float*)alloc((size_t)MROWS * PD * 4);
    float*  Ybu   = (float*)alloc((size_t)MROWS * PD * 4);
    float*  Yparts= (float*)alloc((size_t)NSPLIT * MROWS * PD * 4); // shared by both GEMMs

    k_init<<<BB * P, D, 0, stream>>>(x, cw, cb, fw, fb, cols0);

    const int gemmBlocks = PD / 16;      // 800 blocks x 8 waves = 6400 waves
    for (int step = 0; step < 3; ++step) {
        float* cur = (step & 1) ? cols1 : cols0;
        float* nxt = (step & 1) ? cols0 : cols1;
        int n1 = E * BB * P * D;
        k_gconv<<<(n1 + 255) / 256, 256, 0, stream>>>(cur, qw, qb, kw, kb, vw, vb,
                                                      qbuf, kbuf, vbuf);
        k_scores<<<E * BB, 256, 0, stream>>>(qbuf, kbuf, Sbuf);
        k_softmax<<<E * BB * D, D, 0, stream>>>(Sbuf);
        k_attnout<<<E * BB * P, D, 0, stream>>>(vbuf, Sbuf, cur, gamma, tabuf);
        k_pack_td<<<(MROWS * KTDP + 255) / 256, 256, 0, stream>>>(cur, Xtd);
        k_pack_bu<<<(MROWS * PD + 255) / 256, 256, 0, stream>>>(cur, Xbu);

        k_gemm64<<<gemmBlocks, 256, 0, stream>>>(Xtd, tdw, Yparts, KTDP, KTD, PD);
        k_reduce<<<(MROWS * PD + 255) / 256, 256, 0, stream>>>(Yparts, tdb, Ytd, PD);
        k_gemm64<<<gemmBlocks, 256, 0, stream>>>(Xbu, buw, Yparts, PD, PD, PD);
        k_reduce<<<(MROWS * PD + 255) / 256, 256, 0, stream>>>(Yparts, bub, Ybu, PD);

        k_combine<<<(BPED + 255) / 256, 256, 0, stream>>>(cur, tabuf, Ytd, Ybu, nxt);
    }
    // steps write cols1, cols0, cols1 -> final state in cols1
    k_final<<<(BB * D + 255) / 256, 256, 0, stream>>>(cols1, (float*)d_out);
}
